// SmurfingHunter_85796266705369
// MI455X (gfx1250) — matrix operand order repaired and checked by reading, not compiled, not run on hardware
//
#include <hip/hip_runtime.h>

typedef __attribute__((ext_vector_type(16))) _Float16 v16h;
typedef __attribute__((ext_vector_type(8)))  float    v8f;

#define N_NODES  100000
#define IN_DIM   64
#define HID      128
#define HID2     64
#define N_EDGES  1600000
#define BN_EPS   1e-5f

// ---------------------------------------------------------------- utilities

__global__ void k_zero4(float4* __restrict__ p, int n4) {
  int stride = gridDim.x * blockDim.x;
  for (int i = blockIdx.x * blockDim.x + threadIdx.x; i < n4; i += stride) {
    p[i] = make_float4(0.f, 0.f, 0.f, 0.f);
  }
}

__global__ void k_count(const int* __restrict__ dst, float* __restrict__ cnt, int E) {
  int stride = gridDim.x * blockDim.x;
  for (int e = blockIdx.x * blockDim.x + threadIdx.x; e < E; e += stride) {
    atomicAdd(&cnt[dst[e]], 1.0f);
  }
}

__global__ void k_inv(float* __restrict__ cnt, int n) {
  int i = blockIdx.x * blockDim.x + threadIdx.x;
  if (i < n) cnt[i] = 1.0f / fmaxf(cnt[i], 1.0f);
}

// scatter-add: out[dst] += feat[src] for D = D4*4 features, float4 granularity
__global__ void k_scatter(const float4* __restrict__ feat,
                          const int* __restrict__ src, const int* __restrict__ dst,
                          float* __restrict__ out, int E, int D4, int shift) {
  int total = E * D4;              // <= 51.2M, fits int
  int stride = gridDim.x * blockDim.x;
  for (int t = blockIdx.x * blockDim.x + threadIdx.x; t < total; t += stride) {
    int e = t >> shift;
    int q = t & (D4 - 1);
    int s = src[e];
    int d = dst[e];
    float4 v = feat[s * D4 + q];
    float* o = out + ((long)d * D4 + q) * 4;
    atomicAdd(o + 0, v.x);
    atomicAdd(o + 1, v.y);
    atomicAdd(o + 2, v.z);
    atomicAdd(o + 3, v.w);
  }
}

// ------------------------------------------------------- weight pre-swizzle
// Pack [W_l ; W_r] into f16 WMMA-B fragment order:
//   B1: K=128,N=128 -> [ct(8)][kstep(4)][lane(32)][e(16)]
//   B2: K=256,N=64  -> [ct(4)][kstep(8)][lane(32)][e(16)]
// lane = (kk>=16 ? 16:0) + (n&15), e = kk&15, kk = k&31.

__global__ void k_prepw(const float* __restrict__ W1l, const float* __restrict__ W1r,
                        const float* __restrict__ W2l, const float* __restrict__ W2r,
                        _Float16* __restrict__ B1, _Float16* __restrict__ B2) {
  int t = blockIdx.x * blockDim.x + threadIdx.x;
  if (t < 128 * 128) {
    int k = t >> 7, n = t & 127;
    float v = (k < 64) ? W1l[k * 128 + n] : W1r[(k - 64) * 128 + n];
    int ct = n >> 4, nl = n & 15, kstep = k >> 5, kk = k & 31;
    int lane = ((kk >> 4) << 4) + nl;
    B1[((ct * 4 + kstep) * 32 + lane) * 16 + (kk & 15)] = (_Float16)v;
  } else if (t < 2 * 128 * 128) {
    int u = t - 128 * 128;
    int k = u >> 6, n = u & 63;
    float v = (k < 128) ? W2l[k * 64 + n] : W2r[(k - 128) * 64 + n];
    int ct = n >> 4, nl = n & 15, kstep = k >> 5, kk = k & 31;
    int lane = ((kk >> 4) << 4) + nl;
    B2[((ct * 8 + kstep) * 32 + lane) * 16 + (kk & 15)] = (_Float16)v;
  }
}

// A-fragment swizzle helper: element (m, kk) of a 16x32 f16 A tile lives at
// lane = hi*16 + m, slot e, with hi=kk>>4, e=kk&15.
__device__ __forceinline__ int a_swz(int m, int kk) {
  int hi = (kk >> 4) & 1;
  int e  = kk & 15;
  return (hi * 16 + m) * 16 + e;
}

// ------------------------------------------------- fused SAGE layer 1 GEMM
// h1 = relu(bn1( [mean|x](100k x 128) @ B1(128x128) + b1 ))
// 256 threads (8 waves), 16 rows/block, wave w -> 16-col tile w.

__global__ __launch_bounds__(256) void k_gemm1(
    const float* __restrict__ x, const float* __restrict__ sum1,
    const float* __restrict__ inv, const _Float16* __restrict__ Bh,
    const float* __restrict__ b1,
    const float* __restrict__ g1, const float* __restrict__ bt1,
    const float* __restrict__ mu1, const float* __restrict__ var1,
    float* __restrict__ h1) {
  __shared__ __align__(32) _Float16 sB[128 * 128];  // 32 KB, fragment order
  __shared__ __align__(32) _Float16 sA[16 * 128];   // 4 KB,  fragment order
  const int tid = threadIdx.x;
  const int r0 = blockIdx.x * 16;

  // async byte-copy of pre-swizzled B into LDS (8 x b128 per thread)
  {
    unsigned lbase = (unsigned)(size_t)(void*)&sB[0];
    unsigned long long gbase = (unsigned long long)Bh;
#pragma unroll
    for (int i = 0; i < 8; ++i) {
      unsigned off = (unsigned)tid * 16u + (unsigned)i * 4096u;
      unsigned laddr = lbase + off;
      asm volatile("global_load_async_to_lds_b128 %0, %1, %2"
                   :: "v"(laddr), "v"(off), "s"(gbase) : "memory");
    }
  }
  // stage A (convert f32 -> f16, swizzle to fragment order)
  for (int idx = tid; idx < 16 * 128; idx += 256) {
    int m = idx >> 7, k = idx & 127;
    int row = r0 + m;
    float v = (k < 64) ? sum1[row * 64 + k] * inv[row]
                       : x[row * 64 + (k - 64)];
    sA[(k >> 5) * 512 + a_swz(m, k & 31)] = (_Float16)v;
  }
  asm volatile("s_wait_asynccnt 0" ::: "memory");
  __syncthreads();

  const int lane = tid & 31;
  const int ct   = tid >> 5;

  v8f acc = {0.f, 0.f, 0.f, 0.f, 0.f, 0.f, 0.f, 0.f};
#pragma unroll
  for (int kstep = 0; kstep < 4; ++kstep) {
    v16h a = *(const v16h*)(sA + (kstep * 32 + lane) * 16);
    v16h b = *(const v16h*)(sB + ((ct * 4 + kstep) * 32 + lane) * 16);
    acc = __builtin_amdgcn_wmma_f32_16x16x32_f16(false, a, false, b,
                                                 (short)0, acc, false, false);
  }

  const int hi = lane >> 4, ml = lane & 15;
  const int n  = ct * 16 + ml;
  const float sc   = g1[n] * rsqrtf(var1[n] + BN_EPS);
  const float bias = b1[n];
  const float mean = mu1[n];
  const float beta = bt1[n];
#pragma unroll
  for (int r = 0; r < 8; ++r) {
    int m = r + hi * 8;
    float v = acc[r] + bias;
    v = (v - mean) * sc + beta;
    v = fmaxf(v, 0.f);
    h1[(r0 + m) * 128 + n] = v;
  }
}

// ------------------------------------------------- fused SAGE layer 2 GEMM
// h2 = relu(bn2( [mean2|h1](100k x 256) @ B2(256x64) + b2 ))
// 256 threads (8 waves), 32 rows/block: wave w -> row tile w>>2, col tile w&3.

__global__ __launch_bounds__(256) void k_gemm2(
    const float* __restrict__ h1, const float* __restrict__ sum2,
    const float* __restrict__ inv, const _Float16* __restrict__ Bh,
    const float* __restrict__ b2,
    const float* __restrict__ g2, const float* __restrict__ bt2,
    const float* __restrict__ mu2, const float* __restrict__ var2,
    float* __restrict__ h2) {
  __shared__ __align__(32) _Float16 sB[256 * 64];   // 32 KB, fragment order
  __shared__ __align__(32) _Float16 sA[32 * 256];   // 16 KB, fragment order
  const int tid = threadIdx.x;
  const int r0 = blockIdx.x * 32;

  {
    unsigned lbase = (unsigned)(size_t)(void*)&sB[0];
    unsigned long long gbase = (unsigned long long)Bh;
#pragma unroll
    for (int i = 0; i < 8; ++i) {
      unsigned off = (unsigned)tid * 16u + (unsigned)i * 4096u;
      unsigned laddr = lbase + off;
      asm volatile("global_load_async_to_lds_b128 %0, %1, %2"
                   :: "v"(laddr), "v"(off), "s"(gbase) : "memory");
    }
  }
  for (int idx = tid; idx < 32 * 256; idx += 256) {
    int m = idx >> 8, k = idx & 255;
    int row = r0 + m;
    float v = (k < 128) ? sum2[row * 128 + k] * inv[row]
                        : h1[row * 128 + (k - 128)];
    int rt = m >> 4, ml = m & 15;
    sA[(rt * 8 + (k >> 5)) * 512 + a_swz(ml, k & 31)] = (_Float16)v;
  }
  asm volatile("s_wait_asynccnt 0" ::: "memory");
  __syncthreads();

  const int lane = tid & 31;
  const int w    = tid >> 5;
  const int ct   = w & 3;
  const int rt   = w >> 2;

  v8f acc = {0.f, 0.f, 0.f, 0.f, 0.f, 0.f, 0.f, 0.f};
#pragma unroll
  for (int kstep = 0; kstep < 8; ++kstep) {
    v16h a = *(const v16h*)(sA + ((rt * 8 + kstep) * 32 + lane) * 16);
    v16h b = *(const v16h*)(sB + ((ct * 8 + kstep) * 32 + lane) * 16);
    acc = __builtin_amdgcn_wmma_f32_16x16x32_f16(false, a, false, b,
                                                 (short)0, acc, false, false);
  }

  const int hi = lane >> 4, ml = lane & 15;
  const int n  = ct * 16 + ml;
  const float sc   = g2[n] * rsqrtf(var2[n] + BN_EPS);
  const float bias = b2[n];
  const float mean = mu2[n];
  const float beta = bt2[n];
#pragma unroll
  for (int r = 0; r < 8; ++r) {
    int m = rt * 16 + r + hi * 8;
    float v = acc[r] + bias;
    v = (v - mean) * sc + beta;
    v = fmaxf(v, 0.f);
    h2[(r0 + m) * 64 + n] = v;
  }
}

// ---------------------------------------------------------------- classifier
__global__ void k_cls(const float* __restrict__ h2, const float* __restrict__ Wc,
                      const float* __restrict__ bc, float* __restrict__ out) {
  int i = blockIdx.x * blockDim.x + threadIdx.x;
  if (i >= N_NODES) return;
  const float* hr = h2 + i * 64;
  float a0 = bc[0], a1 = bc[1];
#pragma unroll
  for (int k = 0; k < 64; ++k) {
    float v = hr[k];
    a0 = fmaf(v, Wc[k * 2 + 0], a0);
    a1 = fmaf(v, Wc[k * 2 + 1], a1);
  }
  out[i * 2 + 0] = a0;
  out[i * 2 + 1] = a1;
}

// ---------------------------------------------------------------- launcher

extern "C" void kernel_launch(void* const* d_in, const int* in_sizes, int n_in,
                              void* d_out, int out_size, void* d_ws, size_t ws_size,
                              hipStream_t stream) {
  const float* x    = (const float*)d_in[0];
  const int*   ei   = (const int*)d_in[1];
  const int*   src  = ei;
  const int*   dst  = ei + N_EDGES;
  const float* W1l  = (const float*)d_in[2];
  const float* b1   = (const float*)d_in[3];
  const float* W1r  = (const float*)d_in[4];
  const float* W2l  = (const float*)d_in[5];
  const float* b2   = (const float*)d_in[6];
  const float* W2r  = (const float*)d_in[7];
  const float* g1   = (const float*)d_in[8];
  const float* bt1  = (const float*)d_in[9];
  const float* mu1  = (const float*)d_in[10];
  const float* var1 = (const float*)d_in[11];
  const float* g2   = (const float*)d_in[12];
  const float* bt2  = (const float*)d_in[13];
  const float* mu2  = (const float*)d_in[14];
  const float* var2 = (const float*)d_in[15];
  const float* Wc   = (const float*)d_in[16];
  const float* bc   = (const float*)d_in[17];

  float* ws   = (float*)d_ws;
  float* inv  = ws;                              // N
  float* sum1 = inv  + N_NODES;                  // N*64
  float* sum2 = sum1 + (long)N_NODES * 64;       // N*128
  float* h1   = sum2 + (long)N_NODES * 128;      // N*128
  float* h2   = h1   + (long)N_NODES * 128;      // N*64
  _Float16* B1h = (_Float16*)(h2 + (long)N_NODES * 64);  // 16384 halfs
  _Float16* B2h = B1h + 128 * 128;                       // 16384 halfs

  // pre-swizzle weights (independent of graph work)
  k_prepw<<<128, 256, 0, stream>>>(W1l, W1r, W2l, W2r, B1h, B2h);

  // zero inv + sum1 + sum2 (contiguous, N*193 floats, multiple of 4)
  int zn4 = (N_NODES / 4) * 193;
  k_zero4<<<2048, 256, 0, stream>>>((float4*)ws, zn4);

  k_count<<<2048, 256, 0, stream>>>(dst, inv, N_EDGES);
  k_inv<<<(N_NODES + 255) / 256, 256, 0, stream>>>(inv, N_NODES);

  // layer 1: scatter x (D=64 -> D4=16, shift=4)
  k_scatter<<<4096, 256, 0, stream>>>((const float4*)x, src, dst, sum1,
                                      N_EDGES, 16, 4);
  k_gemm1<<<N_NODES / 16, 256, 0, stream>>>(x, sum1, inv, B1h, b1,
                                            g1, bt1, mu1, var1, h1);

  // layer 2: scatter h1 (D=128 -> D4=32, shift=5)
  k_scatter<<<8192, 256, 0, stream>>>((const float4*)h1, src, dst, sum2,
                                      N_EDGES, 32, 5);
  k_gemm2<<<N_NODES / 32, 256, 0, stream>>>(h1, sum2, inv, B2h, b2,
                                            g2, bt2, mu2, var2, h2);

  k_cls<<<(N_NODES + 255) / 256, 256, 0, stream>>>(h2, Wc, bc, (float*)d_out);
}
